// FeedForward_42717744726585
// MI455X (gfx1250) — compile-verified, matrix operand order checked by reading, not run on hardware
//
#include <hip/hip_runtime.h>

#define SEQ 4096
#define DIM 4096
#define HID 11008

typedef __attribute__((ext_vector_type(16))) __bf16         v16bf;
typedef __attribute__((ext_vector_type(8)))  float          v8f;
typedef __attribute__((ext_vector_type(4)))  unsigned int   u32x4;
typedef __attribute__((ext_vector_type(4)))  float          f32x4;
typedef __attribute__((ext_vector_type(4)))  unsigned short u16x4;

struct bfrag_bits { u32x4 lo, hi; };

// fp32 -> bf16, round-to-nearest-even
static __device__ __forceinline__ unsigned short f2bf(float x) {
    unsigned int u = __float_as_uint(x);
    u += 0x7FFFu + ((u >> 16) & 1u);
    return (unsigned short)(u >> 16);
}

// Build a 16x(bf16) WMMA fragment from two 16-byte LDS chunks.
static __device__ __forceinline__ v16bf make_frag(const unsigned short* p0) {
    bfrag_bits b;
    b.lo = *(const u32x4*)(p0);
    b.hi = *(const u32x4*)(p0 + 16);
    return __builtin_bit_cast(v16bf, b);
}

// Generic pointer to LDS: low 32 bits of the flat address ARE the LDS offset.
static __device__ __forceinline__ unsigned lds_off(const void* p) {
    return (unsigned)(unsigned long long)p;
}

// CDNA5 async copy global -> LDS (ASYNCcnt-tracked, no VGPR staging).
// GVS addressing: mem = SADDR(64) + VADDR(32) + inst_offset.
static __device__ __forceinline__ void async_ld_b128(unsigned lds_addr,
                                                     const unsigned short* base,
                                                     unsigned byte_off) {
    asm volatile("global_load_async_to_lds_b128 %0, %1, %2"
                 :: "v"(lds_addr), "v"(byte_off), "s"(base)
                 : "memory");
}

template <int N>
static __device__ __forceinline__ void wait_asynccnt() {
    asm volatile("s_wait_asynccnt %0" :: "n"(N) : "memory");
}

// ---------------------------------------------------------------------------
// Elementwise fp32 -> bf16 (for x)
// ---------------------------------------------------------------------------
__global__ __launch_bounds__(256) void k_cvt(const float* __restrict__ src,
                                             unsigned short* __restrict__ dst,
                                             int n) {
    int i = (blockIdx.x * 256 + threadIdx.x) * 4;
    if (i + 3 < n) {
        f32x4 v = *(const f32x4*)&src[i];
        u16x4 o;
        o.x = f2bf(v.x); o.y = f2bf(v.y); o.z = f2bf(v.z); o.w = f2bf(v.w);
        *(u16x4*)&dst[i] = o;
    }
}

// ---------------------------------------------------------------------------
// Transpose + convert: src fp32 [K][N] -> dst bf16 [N][K]
// ---------------------------------------------------------------------------
__global__ __launch_bounds__(256) void k_tconv(const float* __restrict__ src,
                                               unsigned short* __restrict__ dst,
                                               int K, int N) {
    __shared__ float t[32][33];
    const int n0 = blockIdx.x * 32;
    const int k0 = blockIdx.y * 32;
    const int tid = threadIdx.x;
    #pragma unroll
    for (int p = 0; p < 4; ++p) {
        int idx = tid + p * 256;
        int r = idx >> 5, c = idx & 31;
        t[r][c] = src[(size_t)(k0 + r) * N + (n0 + c)];
    }
    __syncthreads();
    #pragma unroll
    for (int p = 0; p < 4; ++p) {
        int idx = tid + p * 256;
        int r = idx >> 5, c = idx & 31;
        dst[(size_t)(n0 + r) * K + (k0 + c)] = f2bf(t[c][r]);
    }
}

// ---------------------------------------------------------------------------
// Fused gate/up GEMM: h = silu(x@W1) * (x@W3), bf16 in / bf16 out
// Block tile 128x64, BK=32, 8 waves, wave tile 32x32 per weight.
// Double-buffered LDS filled by async-to-LDS copies.
// ---------------------------------------------------------------------------
#define LDA 40   // padded LDS row stride (elements) to spread banks

__global__ __launch_bounds__(256) void k_gate_up(
    const unsigned short* __restrict__ xb,   // [SEQ][DIM]
    const unsigned short* __restrict__ w1t,  // [HID][DIM]
    const unsigned short* __restrict__ w3t,  // [HID][DIM]
    unsigned short* __restrict__ hout) {     // [SEQ][HID]

    __shared__ __align__(16) unsigned short As[2][128 * LDA];
    __shared__ __align__(16) unsigned short B1[2][64 * LDA];
    __shared__ __align__(16) unsigned short B3[2][64 * LDA];

    const int tid     = threadIdx.x;
    const int m_block = blockIdx.x * 128;
    const int n_block = blockIdx.y * 64;

    const int wave = tid >> 5;
    const int lane = tid & 31;
    const int m0   = (wave & 3) * 32;
    const int n0   = (wave >> 2) * 32;
    const int lr   = lane & 15;
    const int lhi  = lane >> 4;

    // per-thread copy slots
    const int ar0 = tid >> 2,         ao0 = (tid & 3) * 8;          // A chunk 0
    const int ar1 = (tid + 256) >> 2, ao1 = ((tid + 256) & 3) * 8;  // A chunk 1
    const int br  = tid >> 2,         bo  = (tid & 3) * 8;          // B chunk

    v8f accg[2][2], accu[2][2];
    #pragma unroll
    for (int i = 0; i < 2; ++i)
        #pragma unroll
        for (int j = 0; j < 2; ++j) { accg[i][j] = v8f{}; accu[i][j] = v8f{}; }

    // 4 async loads per thread per K-tile
    auto prefetch = [&](int buf, int kt) {
        const unsigned kb = (unsigned)kt * 32u;
        async_ld_b128(lds_off(&As[buf][ar0 * LDA + ao0]), xb,
                      ((unsigned)(m_block + ar0) * DIM + kb + ao0) * 2u);
        async_ld_b128(lds_off(&As[buf][ar1 * LDA + ao1]), xb,
                      ((unsigned)(m_block + ar1) * DIM + kb + ao1) * 2u);
        async_ld_b128(lds_off(&B1[buf][br * LDA + bo]), w1t,
                      ((unsigned)(n_block + br) * DIM + kb + bo) * 2u);
        async_ld_b128(lds_off(&B3[buf][br * LDA + bo]), w3t,
                      ((unsigned)(n_block + br) * DIM + kb + bo) * 2u);
    };

    const int NT = DIM / 32;
    prefetch(0, 0);

    for (int kt = 0; kt < NT; ++kt) {
        const int cur = kt & 1;
        if (kt + 1 < NT) {
            prefetch(cur ^ 1, kt + 1);
            wait_asynccnt<4>();   // tile-kt loads (older, in-order) are done
        } else {
            wait_asynccnt<0>();
        }
        __syncthreads();          // all waves' tile-kt data now in LDS

        v16bf a[2], b1f[2], b3f[2];
        #pragma unroll
        for (int i = 0; i < 2; ++i)
            a[i] = make_frag(&As[cur][(m0 + i * 16 + lr) * LDA + lhi * 8]);
        #pragma unroll
        for (int j = 0; j < 2; ++j) {
            b1f[j] = make_frag(&B1[cur][(n0 + j * 16 + lr) * LDA + lhi * 8]);
            b3f[j] = make_frag(&B3[cur][(n0 + j * 16 + lr) * LDA + lhi * 8]);
        }

        #pragma unroll
        for (int i = 0; i < 2; ++i)
            #pragma unroll
            for (int j = 0; j < 2; ++j) {
                accg[i][j] = __builtin_amdgcn_wmma_f32_16x16x32_bf16(
                    false, a[i], false, b1f[j], (short)0, accg[i][j], false, false);
                accu[i][j] = __builtin_amdgcn_wmma_f32_16x16x32_bf16(
                    false, a[i], false, b3f[j], (short)0, accu[i][j], false, false);
            }
        __syncthreads();          // done reading buf[cur] before it is re-filled
    }

    // ---- epilogue: silu(gate) * up -> bf16 (fast rcp, no IEEE div ladder)
    #pragma unroll
    for (int i = 0; i < 2; ++i)
        #pragma unroll
        for (int j = 0; j < 2; ++j)
            #pragma unroll
            for (int v = 0; v < 8; ++v) {
                int row = m_block + m0 + i * 16 + v + lhi * 8;
                int col = n_block + n0 + j * 16 + lr;
                float g = accg[i][j][v];
                float u = accu[i][j][v];
                float s = g * __builtin_amdgcn_rcpf(1.0f + __expf(-g));
                hout[(size_t)row * HID + col] = f2bf(s * u);
            }
}

// ---------------------------------------------------------------------------
// Down GEMM: out = h @ W2, fp32 out. Same pipeline, single B stream.
// ---------------------------------------------------------------------------
__global__ __launch_bounds__(256) void k_down(
    const unsigned short* __restrict__ hb,   // [SEQ][HID]
    const unsigned short* __restrict__ w2t,  // [DIM][HID]
    float* __restrict__ out) {               // [SEQ][DIM]

    __shared__ __align__(16) unsigned short As[2][128 * LDA];
    __shared__ __align__(16) unsigned short Bs[2][64 * LDA];

    const int tid     = threadIdx.x;
    const int m_block = blockIdx.x * 128;
    const int n_block = blockIdx.y * 64;

    const int wave = tid >> 5;
    const int lane = tid & 31;
    const int m0   = (wave & 3) * 32;
    const int n0   = (wave >> 2) * 32;
    const int lr   = lane & 15;
    const int lhi  = lane >> 4;

    const int ar0 = tid >> 2,         ao0 = (tid & 3) * 8;
    const int ar1 = (tid + 256) >> 2, ao1 = ((tid + 256) & 3) * 8;
    const int br  = tid >> 2,         bo  = (tid & 3) * 8;

    v8f acc[2][2];
    #pragma unroll
    for (int i = 0; i < 2; ++i)
        #pragma unroll
        for (int j = 0; j < 2; ++j) acc[i][j] = v8f{};

    auto prefetch = [&](int buf, int kt) {
        const unsigned kb = (unsigned)kt * 32u;
        async_ld_b128(lds_off(&As[buf][ar0 * LDA + ao0]), hb,
                      ((unsigned)(m_block + ar0) * HID + kb + ao0) * 2u);
        async_ld_b128(lds_off(&As[buf][ar1 * LDA + ao1]), hb,
                      ((unsigned)(m_block + ar1) * HID + kb + ao1) * 2u);
        async_ld_b128(lds_off(&Bs[buf][br * LDA + bo]), w2t,
                      ((unsigned)(n_block + br) * HID + kb + bo) * 2u);
    };

    const int NT = HID / 32;
    prefetch(0, 0);

    for (int kt = 0; kt < NT; ++kt) {
        const int cur = kt & 1;
        if (kt + 1 < NT) {
            prefetch(cur ^ 1, kt + 1);
            wait_asynccnt<3>();
        } else {
            wait_asynccnt<0>();
        }
        __syncthreads();

        v16bf a[2], b[2];
        #pragma unroll
        for (int i = 0; i < 2; ++i)
            a[i] = make_frag(&As[cur][(m0 + i * 16 + lr) * LDA + lhi * 8]);
        #pragma unroll
        for (int j = 0; j < 2; ++j)
            b[j] = make_frag(&Bs[cur][(n0 + j * 16 + lr) * LDA + lhi * 8]);

        #pragma unroll
        for (int i = 0; i < 2; ++i)
            #pragma unroll
            for (int j = 0; j < 2; ++j)
                acc[i][j] = __builtin_amdgcn_wmma_f32_16x16x32_bf16(
                    false, a[i], false, b[j], (short)0, acc[i][j], false, false);
        __syncthreads();
    }

    #pragma unroll
    for (int i = 0; i < 2; ++i)
        #pragma unroll
        for (int j = 0; j < 2; ++j)
            #pragma unroll
            for (int v = 0; v < 8; ++v) {
                int row = m_block + m0 + i * 16 + v + lhi * 8;
                int col = n_block + n0 + j * 16 + lr;
                out[(size_t)row * DIM + col] = acc[i][j][v];
            }
}

// ---------------------------------------------------------------------------
extern "C" void kernel_launch(void* const* d_in, const int* in_sizes, int n_in,
                              void* d_out, int out_size, void* d_ws, size_t ws_size,
                              hipStream_t stream) {
    const float* x  = (const float*)d_in[0];   // [SEQ][DIM]
    const float* w1 = (const float*)d_in[1];   // [DIM][HID]
    const float* w2 = (const float*)d_in[2];   // [HID][DIM]
    const float* w3 = (const float*)d_in[3];   // [DIM][HID]

    char*  ws  = (char*)d_ws;
    size_t off = 0;
    auto alloc = [&](size_t bytes) -> void* {
        void* p = ws + off;
        off += (bytes + 255) & ~(size_t)255;
        return p;
    };
    unsigned short* xb  = (unsigned short*)alloc((size_t)SEQ * DIM * 2);
    unsigned short* w1t = (unsigned short*)alloc((size_t)HID * DIM * 2);
    unsigned short* w3t = (unsigned short*)alloc((size_t)HID * DIM * 2);
    unsigned short* w2t = (unsigned short*)alloc((size_t)DIM * HID * 2);
    unsigned short* hb  = (unsigned short*)alloc((size_t)SEQ * HID * 2);

    k_cvt<<<(SEQ * DIM) / 1024, 256, 0, stream>>>(x, xb, SEQ * DIM);
    {
        dim3 g(HID / 32, DIM / 32);          // W1,W3: [DIM][HID] -> [HID][DIM]
        k_tconv<<<g, 256, 0, stream>>>(w1, w1t, DIM, HID);
        k_tconv<<<g, 256, 0, stream>>>(w3, w3t, DIM, HID);
    }
    {
        dim3 g(DIM / 32, HID / 32);          // W2: [HID][DIM] -> [DIM][HID]
        k_tconv<<<g, 256, 0, stream>>>(w2, w2t, HID, DIM);
    }
    {
        dim3 g(SEQ / 128, HID / 64);
        k_gate_up<<<g, 256, 0, stream>>>(xb, w1t, w3t, hb);
    }
    {
        dim3 g(SEQ / 128, DIM / 64);
        k_down<<<g, 256, 0, stream>>>(hb, w2t, (float*)d_out);
    }
}
